// BiGCN_63582695850940
// MI455X (gfx1250) — compile-verified, matrix operand order checked by reading, not run on hardware
//
#include <hip/hip_runtime.h>
#include <hip/hip_bf16.h>

// ---------------------------------------------------------------------------
// Problem dims (fixed by reference)
// ---------------------------------------------------------------------------
#define NN     10000
#define IN_C   512
#define HID    256
#define MID    128
#define OUT_C  64

typedef __attribute__((ext_vector_type(16))) __bf16 v16bf;
typedef __attribute__((ext_vector_type(8)))  __bf16 v8bf;
typedef __attribute__((ext_vector_type(4)))  __bf16 v4bf;
typedef __attribute__((ext_vector_type(8)))  float  v8f;

// 16-byte int vector type matching the async-LDS builtin's parameter type
typedef int v4i_ __attribute__((vector_size(4 * sizeof(int))));
typedef __attribute__((address_space(1))) v4i_ gv4i;   // global
typedef __attribute__((address_space(3))) v4i_ lv4i;   // LDS

// CDNA5 async global->LDS path (guarded; falls back to sync staging)
#if defined(__has_builtin)
#if __has_builtin(__builtin_amdgcn_global_load_async_to_lds_b128) && \
    __has_builtin(__builtin_amdgcn_s_wait_asynccnt)
#define USE_ASYNC_LDS 1
#endif
#endif
#ifndef USE_ASYNC_LDS
#define USE_ASYNC_LDS 0
#endif

// ---------------------------------------------------------------------------
// Column mean / rstd over the batch dim (BatchNorm1d, training, affine=False)
// ---------------------------------------------------------------------------
__global__ void colstats_kernel(const float* __restrict__ x,
                                float* __restrict__ mean,
                                float* __restrict__ rstd,
                                int n, int c) {
  const int col = blockIdx.x;
  float s = 0.f, s2 = 0.f;
  for (int i = threadIdx.x; i < n; i += blockDim.x) {
    float v = x[(size_t)i * c + col];
    s += v; s2 += v * v;
  }
  __shared__ float r0[256];
  __shared__ float r1[256];
  r0[threadIdx.x] = s;
  r1[threadIdx.x] = s2;
  __syncthreads();
  for (int off = 128; off > 0; off >>= 1) {
    if ((int)threadIdx.x < off) {
      r0[threadIdx.x] += r0[threadIdx.x + off];
      r1[threadIdx.x] += r1[threadIdx.x + off];
    }
    __syncthreads();
  }
  if (threadIdx.x == 0) {
    float m   = r0[0] / (float)n;
    float var = r1[0] / (float)n - m * m;
    mean[col] = m;
    rstd[col] = rsqrtf(var + 1e-5f);
  }
}

// Apply BN and emit bf16
__global__ void bn_apply_kernel(const float* __restrict__ x,
                                const float* __restrict__ mean,
                                const float* __restrict__ rstd,
                                __bf16* __restrict__ out,
                                int n, int c) {
  size_t i = (size_t)blockIdx.x * blockDim.x + threadIdx.x;
  if (i >= (size_t)n * c) return;
  int col = (int)(i % (size_t)c);
  out[i] = (__bf16)((x[i] - mean[col]) * rstd[col]);
}

// fp32 -> bf16, 4 elements/thread, grid-stride
__global__ void f32_to_bf16_vec4(const float* __restrict__ in,
                                 __bf16* __restrict__ out, size_t n4) {
  size_t stride = (size_t)gridDim.x * blockDim.x;
  for (size_t i = (size_t)blockIdx.x * blockDim.x + threadIdx.x; i < n4; i += stride) {
    float4 v = ((const float4*)in)[i];
    v4bf o;
    o[0] = (__bf16)v.x; o[1] = (__bf16)v.y;
    o[2] = (__bf16)v.z; o[3] = (__bf16)v.w;
    ((v4bf*)out)[i] = o;
  }
}

// fp32 (R x C, row-major) -> bf16 transposed (C x R): weights as col-major B
__global__ void f32_to_bf16_transpose(const float* __restrict__ in,
                                      __bf16* __restrict__ out, int R, int C) {
  int c = blockIdx.x * 16 + (threadIdx.x & 15);
  int r = blockIdx.y * 16 + (threadIdx.x >> 4);
  if (r < R && c < C)
    out[(size_t)c * R + r] = (__bf16)in[(size_t)r * C + c];
}

// ---------------------------------------------------------------------------
// WMMA GEMM: C[M x Nn] = A[M x K] * B[K x Nn] (+bias)(dropout)
// A: bf16 row-major. B supplied COLUMN-major (Bc[n][k], row stride K) so both
// tiles stage as contiguous 16B chunks -> async global->LDS, double buffered.
// Requires K % 8 == 0 (true for all call sites: 512/10000/256/128).
// Block tile 256x64, 8 waves; each wave: 32-row strip x 64 cols = 8 WMMAs
// sharing 4 B fragments (8 wmma : 12 ds_load_b128 per K-step).
// Staging specialized: interior blocks + full K-steps are branch-free async.
// ---------------------------------------------------------------------------
#define BM 256
#define BN 64
#define BK 32
#define LDST 40   // LDS row stride (bf16 elems) = 80B: bank spread, 16B aligned
#define ATSZ (BM * LDST)
#define BTSZ (BN * LDST)

__device__ __forceinline__ void copy16(__bf16* lds_dst, const __bf16* gsrc) {
#if USE_ASYNC_LDS
  __builtin_amdgcn_global_load_async_to_lds_b128(
      (gv4i*)(size_t)gsrc, (lv4i*)lds_dst, 0, 0);
#else
  *(v8bf*)lds_dst = *(const v8bf*)gsrc;
#endif
}

__device__ __forceinline__ void copy16_or_zero(__bf16* lds_dst,
                                               const __bf16* gsrc, bool valid) {
  if (valid) {
    copy16(lds_dst, gsrc);
  } else {
    v8bf z = {};
    *(v8bf*)lds_dst = z;
  }
}

__device__ __forceinline__ void wait_stage() {
#if USE_ASYNC_LDS
  __builtin_amdgcn_s_wait_asynccnt(0);
#endif
  __syncthreads();
}

__global__ __launch_bounds__(256) void wmma_gemm_bf16(
    const __bf16* __restrict__ A, const __bf16* __restrict__ Bc,
    const float* __restrict__ bias,
    float* __restrict__ outF, __bf16* __restrict__ outB,
    int M, int Nn, int K, unsigned dropSeed, int transB)
{
  __shared__ __attribute__((aligned(16))) __bf16 At[2 * ATSZ];
  __shared__ __attribute__((aligned(16))) __bf16 Bt[2 * BTSZ];

  const int tid  = threadIdx.x;
  const int wave = tid >> 5;
  const int lane = tid & 31;
  const int bm0  = blockIdx.x * BM;
  const int bn0  = blockIdx.y * BN;
  const bool interior = (bm0 + BM) <= M;   // uniform: no row guards needed

  v8f acc[2][4];
  #pragma unroll
  for (int st = 0; st < 2; ++st)
    #pragma unroll
    for (int t = 0; t < 4; ++t)
      #pragma unroll
      for (int j = 0; j < 8; ++j) acc[st][t][j] = 0.0f;

  // Fragment addressing per 16-bit WMMA VGPR layout:
  // lanes 0-15: K chunks [0..7]+[16..23]; lanes 16-31: [8..15]+[24..31]
  const int fr = lane & 15;
  const int kb = (lane >> 4) << 3;   // 0 or 8

  // Per-thread staging coordinates (constant across K-steps)
  const int a0r = tid >> 2;                 // A base row   (0..63), +64*i
  const int aok = (tid & 3) << 3;           // A k-off      {0,8,16,24}
  const int bnr = tid >> 2;                 // B row (n)    (0..63)
  const int bok = (tid & 3) << 3;           // B k-off      {0,8,16,24}

  const __bf16* gA = A  + (size_t)(bm0 + a0r) * K + aok;
  const __bf16* gB = Bc + (size_t)(bn0 + bnr) * K + bok;

  // Branch-free staging: five 16B chunks issued async, no EXEC churn.
  auto stage_fast = [&](int buf, int k0) {
    __bf16* at = At + buf * ATSZ;
    __bf16* bt = Bt + buf * BTSZ;
    #pragma unroll
    for (int i = 0; i < 4; ++i)
      copy16(at + (a0r + i * 64) * LDST + aok, gA + (size_t)(i * 64) * K + k0);
    copy16(bt + bnr * LDST + bok, gB + k0);
  };
  // Edge staging: per-lane bounds, zero-fill OOB (chunk granular; K%8==0).
  auto stage_guard = [&](int buf, int k0) {
    __bf16* at = At + buf * ATSZ;
    __bf16* bt = Bt + buf * BTSZ;
    #pragma unroll
    for (int i = 0; i < 4; ++i)
      copy16_or_zero(at + (a0r + i * 64) * LDST + aok,
                     gA + (size_t)(i * 64) * K + k0,
                     (bm0 + a0r + i * 64 < M) && (k0 + aok < K));
    copy16_or_zero(bt + bnr * LDST + bok, gB + k0, (k0 + bok < K));
  };
  auto stage = [&](int buf, int k0) {
    if (interior && (k0 + BK) <= K) stage_fast(buf, k0);   // uniform select
    else                            stage_guard(buf, k0);
  };

  const int nsteps = (K + BK - 1) / BK;
  stage(0, 0);
  for (int s = 0; s < nsteps; ++s) {
    const int buf = s & 1;
    // publish buffer `buf`; also guarantees everyone is done reading buf^1
    wait_stage();
    if (s + 1 < nsteps) stage(buf ^ 1, (s + 1) * BK);

    const __bf16* at = At + buf * ATSZ;
    const __bf16* bt = Bt + buf * BTSZ;

    // 4 B fragments, shared by both A strips of this wave
    v16bf bfrag[4];
    #pragma unroll
    for (int nt = 0; nt < 4; ++nt) {
      const __bf16* brow = bt + (nt * 16 + fr) * LDST;
      v8bf blo = *(const v8bf*)(brow + kb);
      v8bf bhi = *(const v8bf*)(brow + 16 + kb);
      bfrag[nt] = __builtin_shufflevector(blo, bhi,
          0,1,2,3,4,5,6,7,8,9,10,11,12,13,14,15);
    }
    #pragma unroll
    for (int st = 0; st < 2; ++st) {
      const __bf16* arow = at + (wave * 32 + st * 16 + fr) * LDST;
      v8bf alo = *(const v8bf*)(arow + kb);
      v8bf ahi = *(const v8bf*)(arow + 16 + kb);
      v16bf afrag = __builtin_shufflevector(alo, ahi,
          0,1,2,3,4,5,6,7,8,9,10,11,12,13,14,15);
      #pragma unroll
      for (int nt = 0; nt < 4; ++nt) {
        acc[st][nt] = __builtin_amdgcn_wmma_f32_16x16x32_bf16(
            false, afrag, false, bfrag[nt], (short)0, acc[st][nt],
            false, false);
      }
    }
  }

  // ---- epilogue: bias + dropout + store (f32 and/or bf16, opt transposed) --
  // C layout: VGPR j -> row = j (lanes 0-15) / 8+j (lanes 16-31); col = lane&15
  #pragma unroll
  for (int st = 0; st < 2; ++st) {
    const int rowBase = bm0 + wave * 32 + st * 16 + ((lane >> 4) << 3);
    #pragma unroll
    for (int nt = 0; nt < 4; ++nt) {
      int col = bn0 + nt * 16 + (lane & 15);
      float bv = bias ? bias[col] : 0.0f;
      #pragma unroll
      for (int j = 0; j < 8; ++j) {
        int row = rowBase + j;
        if (!interior && row >= M) continue;   // uniform unswitch on interior
        float v = acc[st][nt][j] + bv;
        if (dropSeed) {
          unsigned h = (unsigned)(row * Nn + col) * 2654435761u + dropSeed;
          h ^= h >> 16; h *= 0x85ebca6bu; h ^= h >> 13;
          v = (h & 1u) ? v * 2.0f : 0.0f;
        }
        if (outF) outF[(size_t)row * Nn + col] = v;
        if (outB) {
          size_t o = transB ? ((size_t)col * M + row) : ((size_t)row * Nn + col);
          outB[o] = (__bf16)v;
        }
      }
    }
  }
}

// ---------------------------------------------------------------------------
// log_softmax over 64 columns; one wave per row
// ---------------------------------------------------------------------------
__global__ void logsoftmax64_kernel(const float* __restrict__ in,
                                    float* __restrict__ out, int M) {
  int row  = blockIdx.x * 8 + (threadIdx.x >> 5);
  int lane = threadIdx.x & 31;
  if (row >= M) return;
  const float* p = in + (size_t)row * OUT_C;
  float v0 = p[lane], v1 = p[lane + 32];
  float mx = fmaxf(v0, v1);
  #pragma unroll
  for (int off = 16; off > 0; off >>= 1)
    mx = fmaxf(mx, __shfl_xor(mx, off, 32));
  float e = __expf(v0 - mx) + __expf(v1 - mx);
  #pragma unroll
  for (int off = 16; off > 0; off >>= 1)
    e += __shfl_xor(e, off, 32);
  float lse = mx + __logf(e);
  out[(size_t)row * OUT_C + lane]      = v0 - lse;
  out[(size_t)row * OUT_C + lane + 32] = v1 - lse;
}

// ---------------------------------------------------------------------------
// Host orchestration
// ---------------------------------------------------------------------------
extern "C" void kernel_launch(void* const* d_in, const int* in_sizes, int n_in,
                              void* d_out, int out_size, void* d_ws, size_t ws_size,
                              hipStream_t stream) {
  const float* x   = (const float*)d_in[0];
  const float* adj = (const float*)d_in[1];
  const float* W0  = (const float*)d_in[2];
  const float* b0  = (const float*)d_in[3];
  const float* W1  = (const float*)d_in[4];
  const float* b1  = (const float*)d_in[5];
  const float* W2  = (const float*)d_in[6];
  const float* b2  = (const float*)d_in[7];

  char* ws = (char*)d_ws;
  size_t off = 0;
  auto alloc = [&](size_t bytes) -> void* {
    off = (off + 255) & ~(size_t)255;
    void* p = ws + off;
    off += bytes;
    return p;
  };

  __bf16* adjb = (__bf16*)alloc((size_t)NN * NN * 2);
  __bf16* h0b  = (__bf16*)alloc((size_t)NN * IN_C * 2);
  __bf16* W0t  = (__bf16*)alloc((size_t)IN_C * HID * 2);   // [HID][IN_C]
  __bf16* W1t  = (__bf16*)alloc((size_t)HID * MID * 2);    // [MID][HID]
  __bf16* W2t  = (__bf16*)alloc((size_t)MID * OUT_C * 2);  // [OUT_C][MID]
  __bf16* t0t  = (__bf16*)alloc((size_t)NN * HID * 2);     // (x W0)^T  [HID][NN]
  __bf16* h1b  = (__bf16*)alloc((size_t)NN * HID * 2);     // layer-1 out (row-major)
  __bf16* t1t  = (__bf16*)alloc((size_t)NN * MID * 2);     // (h1 W1)^T [MID][NN]
  __bf16* h2b  = (__bf16*)alloc((size_t)NN * MID * 2);     // layer-2 out (row-major)
  __bf16* t2t  = (__bf16*)alloc((size_t)NN * OUT_C * 2);   // (h2 W2)^T [OUT_C][NN]
  float*  h3f  = (float*)alloc((size_t)NN * OUT_C * 4);    // pre-softmax
  float*  mean = (float*)alloc(IN_C * 4);
  float*  rstd = (float*)alloc(IN_C * 4);

  // 1. BatchNorm stats + apply (bf16 out)
  colstats_kernel<<<IN_C, 256, 0, stream>>>(x, mean, rstd, NN, IN_C);
  {
    size_t n = (size_t)NN * IN_C;
    bn_apply_kernel<<<(unsigned)((n + 255) / 256), 256, 0, stream>>>(
        x, mean, rstd, h0b, NN, IN_C);
  }

  // 2. Convert adj to bf16; weights to bf16 transposed (col-major B operands)
  f32_to_bf16_vec4<<<32768, 256, 0, stream>>>(adj, adjb, (size_t)NN * NN / 4);
  f32_to_bf16_transpose<<<dim3(HID / 16, IN_C / 16), 256, 0, stream>>>(W0, W0t, IN_C, HID);
  f32_to_bf16_transpose<<<dim3(MID / 16, HID / 16), 256, 0, stream>>>(W1, W1t, HID, MID);
  f32_to_bf16_transpose<<<dim3(OUT_C / 16, MID / 16), 256, 0, stream>>>(W2, W2t, MID, OUT_C);

  const int MT = (NN + BM - 1) / BM;  // 40

  // 3. Layer 0: t0 = h0 @ W0 (emit transposed) ; h1 = adj @ t0 + b0, dropout
  wmma_gemm_bf16<<<dim3(MT, HID / BN), 256, 0, stream>>>(
      h0b, W0t, nullptr, nullptr, t0t, NN, HID, IN_C, 0u, 1);
  wmma_gemm_bf16<<<dim3(MT, HID / BN), 256, 0, stream>>>(
      adjb, t0t, b0, nullptr, h1b, NN, HID, NN, 0x9E3779B9u, 0);

  // 4. Layer 1
  wmma_gemm_bf16<<<dim3(MT, MID / BN), 256, 0, stream>>>(
      h1b, W1t, nullptr, nullptr, t1t, NN, MID, HID, 0u, 1);
  wmma_gemm_bf16<<<dim3(MT, MID / BN), 256, 0, stream>>>(
      adjb, t1t, b1, nullptr, h2b, NN, MID, NN, 0x7F4A7C15u, 0);

  // 5. Layer 2 (fp32 out for softmax)
  wmma_gemm_bf16<<<dim3(MT, OUT_C / BN), 256, 0, stream>>>(
      h2b, W2t, nullptr, nullptr, t2t, NN, OUT_C, MID, 0u, 1);
  wmma_gemm_bf16<<<dim3(MT, OUT_C / BN), 256, 0, stream>>>(
      adjb, t2t, b2, h3f, nullptr, NN, OUT_C, NN, 0u, 0);

  // 6. log_softmax
  logsoftmax64_kernel<<<(NN + 7) / 8, 256, 0, stream>>>(h3f, (float*)d_out, NN);
}
